// FlowStateS5Block_2199023255790
// MI455X (gfx1250) — matrix operand order repaired and checked by reading, not compiled, not run on hardware
//
#include <hip/hip_runtime.h>
#include <hip/hip_bf16.h>

// ---------------- problem constants ----------------
constexpr int L_ = 2048, B_ = 32, H_ = 128, P_ = 256;
constexpr size_t M_ = (size_t)L_ * B_;          // 65536 rows (l*B + b)
constexpr int CL_ = 64;                          // scan chunk length
constexpr int CH_ = L_ / CL_;                    // 32 chunks (A^64 = 6 squarings)

typedef __attribute__((ext_vector_type(16))) __bf16 v16bf;
typedef __attribute__((ext_vector_type(8)))  float  v8f;

__device__ inline unsigned short f2bf(float x) {
    union { __bf16 h; unsigned short u; } c;
    c.h = (__bf16)x;
    return c.u;
}

// ---- CDNA5 async global->LDS copy (ASYNCcnt), 16 bytes per lane ----
// LDS address = low 32 bits of the generic pointer.
__device__ inline void async_g2l_b128(const unsigned short* lds_dst, const unsigned short* gsrc) {
    uint32_t lofs = (uint32_t)(uintptr_t)lds_dst;
    uint64_t ga   = (uint64_t)(uintptr_t)gsrc;
    asm volatile("global_load_async_to_lds_b128 %0, %1, off"
                 :: "v"(lofs), "v"(ga) : "memory");
}
__device__ inline void wait_async0() {
    asm volatile("s_wait_asynccnt 0x0" ::: "memory");
}

// ---------------- setup: per-p SSM parameters ----------------
// Apar layout (floats): [0..P) A_r | [P..2P) A_i | [2P..3P) A64_r | [3P..4P) A64_i
//                       [4P..5P) coef_r | [5P..6P) coef_i   (coef = (A-1)/lambda)
__global__ void setup_params(const float* __restrict__ logLr,
                             const float* __restrict__ Limag,
                             const float* __restrict__ logDelta,
                             float* __restrict__ Apar) {
    int p = blockIdx.x * blockDim.x + threadIdx.x;
    if (p >= P_) return;
    float lr = -expf(logLr[p]);
    float li = Limag[p];
    float dt = expf(logDelta[p]);
    float xr = lr * dt, xi = li * dt;           // log(A_bar)
    float er = expf(xr);
    float Ar = er * cosf(xi), Ai = er * sinf(xi);
    // coef = (A - 1) / lambda  (complex divide)
    float nr = Ar - 1.0f, ni = Ai;
    float den = lr * lr + li * li;
    float cr = (nr * lr + ni * li) / den;
    float ci = (ni * lr - nr * li) / den;
    // A^64 by 6 squarings
    float pr = Ar, pi = Ai;
    #pragma unroll
    for (int s = 0; s < 6; ++s) { float qr = pr*pr - pi*pi; float qi = 2.0f*pr*pi; pr = qr; pi = qi; }
    Apar[p]        = Ar;  Apar[P_ + p]     = Ai;
    Apar[2*P_ + p] = pr;  Apar[3*P_ + p]   = pi;
    Apar[4*P_ + p] = cr;  Apar[5*P_ + p]   = ci;
}

// W1t[n][k=h] (512x128 bf16, k contiguous): n<256 -> B_bar_r[p=n,h]; n>=256 -> B_bar_i
__global__ void build_w1(const float* __restrict__ Btr, const float* __restrict__ Bti,
                         const float* __restrict__ Apar, unsigned short* __restrict__ W1t) {
    int idx = blockIdx.x * blockDim.x + threadIdx.x;       // P*H threads
    if (idx >= P_ * H_) return;
    int p = idx / H_, h = idx % H_;                        // h fastest -> coalesced writes
    float cr = Apar[4*P_ + p], ci = Apar[5*P_ + p];
    float br = Btr[p * H_ + h], bi = Bti[p * H_ + h];
    W1t[(size_t)p * H_ + h]          = f2bf(cr * br - ci * bi);
    W1t[(size_t)(256 + p) * H_ + h]  = f2bf(cr * bi + ci * br);
}

// W2t[n=h][k] (128x512 bf16, k contiguous): k<256 -> C_r[h,k]; else -> -C_i[h,k-256]
__global__ void build_w2(const float* __restrict__ Cr, const float* __restrict__ Ci,
                         unsigned short* __restrict__ W2t) {
    int idx = blockIdx.x * blockDim.x + threadIdx.x;       // H*512 threads
    if (idx >= H_ * 512) return;
    int h = idx >> 9, k = idx & 511;                       // k fastest -> coalesced
    float v = (k < 256) ? Cr[h * P_ + k] : -Ci[h * P_ + (k - 256)];
    W2t[(size_t)h * 512 + k] = f2bf(v);
}

__global__ void convert_u(const float4* __restrict__ U4, unsigned short* __restrict__ Ubf, size_t n4) {
    size_t i = (size_t)blockIdx.x * blockDim.x + threadIdx.x;
    if (i >= n4) return;
    float4 v = U4[i];
    union { unsigned short s[4]; uint2 u; } o;
    o.s[0] = f2bf(v.x); o.s[1] = f2bf(v.y); o.s[2] = f2bf(v.z); o.s[3] = f2bf(v.w);
    *(uint2*)&Ubf[i * 4] = o.u;
}

// ---------------- WMMA bf16 GEMM:  C[M,N] = A[M,K] * Wt[N,K]^T (+ optional D*u epilogue) ----
// Block tile 128x128, 256 threads = 8 waves (4 x 2), each wave 32(M) x 64(N) = 8 wmma accs.
// Both operands are staged with async global->LDS b128 copies, double-buffered.
#define BM 128
#define BN 128
#define BK 32

__global__ __launch_bounds__(256)
void gemm_bf16_wmma(const unsigned short* __restrict__ A,    // [M][K] bf16
                    const unsigned short* __restrict__ Wt,   // [N][K] bf16 (pre-transposed)
                    float* __restrict__ Cout,                // [M][N] f32
                    int M, int N, int K,
                    const float* __restrict__ Dvec,          // epilogue if != nullptr
                    const float* __restrict__ Uin) {
    __shared__ unsigned short sA[2][BM * BK];   // [row][k]  8 KB each
    __shared__ unsigned short sB[2][BN * BK];   // [n][k]    8 KB each

    const int tid   = threadIdx.x;
    const int wave  = tid >> 5;
    const int lane  = tid & 31;
    const int wm    = wave >> 1;             // 0..3 : 32-row strip
    const int wn    = wave & 1;              // 0..1 : 64-col strip
    const int m0    = blockIdx.x * BM;
    const int n0    = blockIdx.y * BN;
    const int lhalf = lane >> 4;             // 0 | 1
    const int l15   = lane & 15;

    // per-thread staging slots: both tiles are 128 rows x 64 bytes = 512 x 16B
    const int r_   = tid >> 1;               // wait no -- 2 segs per thread, computed in stage()

    v8f acc[2][4];
    const v8f vzero = {0.f,0.f,0.f,0.f,0.f,0.f,0.f,0.f};
    #pragma unroll
    for (int im = 0; im < 2; ++im)
        #pragma unroll
        for (int in = 0; in < 4; ++in) acc[im][in] = vzero;

    // issue async copies for one 128x32 bf16 tile (4 segs of 16B per row)
    auto stage = [&](const unsigned short* base, int row0, int kb, unsigned short* dst) {
        #pragma unroll
        for (int i = 0; i < 2; ++i) {
            int linear = tid + i * 256;          // 0..511
            int r = linear >> 2, seg = linear & 3;
            async_g2l_b128(&dst[r * BK + seg * 8],
                           &base[(size_t)(row0 + r) * K + kb + seg * 8]);
        }
    };

    const int nk = K / BK;
    stage(A,  m0, 0, sA[0]);
    stage(Wt, n0, 0, sB[0]);

    for (int ib = 0; ib < nk; ++ib) {
        const int cur = ib & 1;
        wait_async0();          // my copies into buf[cur] have landed in LDS
        __syncthreads();        // everyone's copies landed; everyone done reading buf[cur^1]
        if (ib + 1 < nk) {      // overlap next tile's copies with this tile's WMMAs
            stage(A,  m0, (ib + 1) * BK, sA[cur ^ 1]);
            stage(Wt, n0, (ib + 1) * BK, sB[cur ^ 1]);
        }

        // Fragments per ISA layout.
        // A 16x32 bf16: lane = row (mod 16); k(j) = (j>>2)*16 + lhalf*8 + (j&3)*2  (pair)
        // B 32x16 bf16 (n-major LDS): lane = col (mod 16); k(j) = (j>>2)*16 + lhalf*8 + (j&3)*2  (pair)
        union Frag { v16bf v; uint32_t u[8]; };
        Frag fa[2], fb[4];
        #pragma unroll
        for (int im = 0; im < 2; ++im) {
            int mrow = wm * 32 + im * 16 + l15;
            #pragma unroll
            for (int j = 0; j < 8; ++j) {
                int kk = ((j >> 2) * 16) + lhalf * 8 + ((j & 3) * 2);
                fa[im].u[j] = *(const uint32_t*)&sA[cur][mrow * BK + kk];
            }
        }
        #pragma unroll
        for (int in = 0; in < 4; ++in) {
            int ncol = wn * 64 + in * 16 + l15;
            #pragma unroll
            for (int j = 0; j < 8; ++j) {
                int kk = ((j >> 2) * 16) + lhalf * 8 + ((j & 3) * 2);
                fb[in].u[j] = *(const uint32_t*)&sB[cur][ncol * BK + kk];
            }
        }
        #pragma unroll
        for (int im = 0; im < 2; ++im)
            #pragma unroll
            for (int in = 0; in < 4; ++in)
                acc[im][in] = __builtin_amdgcn_wmma_f32_16x16x32_bf16(
                    false, fa[im].v, false, fb[in].v,
                    (short)0, acc[im][in], false, false);
    }

    // C/D layout: lane = col (mod 16); VGPR j = row j + 8*lhalf
    #pragma unroll
    for (int im = 0; im < 2; ++im) {
        #pragma unroll
        for (int in = 0; in < 4; ++in) {
            int nc = n0 + wn * 64 + in * 16 + l15;
            #pragma unroll
            for (int j = 0; j < 8; ++j) {
                int mr = m0 + wm * 32 + im * 16 + lhalf * 8 + j;
                float v = acc[im][in][j];
                if (Dvec) v += Dvec[nc] * Uin[(size_t)mr * N + nc];
                Cout[(size_t)mr * N + nc] = v;
            }
        }
    }
    (void)r_;
}

// ---------------- chunked scan over time ----------------
// Bu layout: row m = t*B + b, 512 floats: [0..256) = Bu_r[p], [256..512) = Bu_i[p]
// S layout: planes: S[idx] (real), S[SOFF + idx] (imag), idx = (c*B + b)*256 + p
constexpr size_t SOFF_ = (size_t)CH_ * B_ * P_;

__global__ __launch_bounds__(256)
void scan_local(const float* __restrict__ Bu, const float* __restrict__ Apar,
                float* __restrict__ S) {
    int p = threadIdx.x;
    int b = blockIdx.x / CH_;
    int c = blockIdx.x % CH_;
    float Ar = Apar[p], Ai = Apar[P_ + p];
    float sr = 0.f, si = 0.f;
    int t0 = c * CL_;
    for (int t = t0; t < t0 + CL_; ++t) {
        const float* row = Bu + ((size_t)t * B_ + b) * 512;
        float br = row[p], bi = row[256 + p];
        float qr = Ar * sr - Ai * si + br;
        float qi = Ar * si + Ai * sr + bi;
        sr = qr; si = qi;
    }
    size_t idx = ((size_t)c * B_ + b) * P_ + p;
    S[idx] = sr; S[SOFF_ + idx] = si;
}

__global__ void scan_combine(const float* __restrict__ Apar, float* __restrict__ S) {
    int tid = blockIdx.x * blockDim.x + threadIdx.x;   // 8192 = B*P
    int p = tid & (P_ - 1), b = tid >> 8;
    float Gr = Apar[2*P_ + p], Gi = Apar[3*P_ + p];    // A^CL
    float ir = 0.f, ii = 0.f;
    for (int c = 0; c < CH_; ++c) {
        size_t idx = ((size_t)c * B_ + b) * P_ + p;
        float tr = S[idx], ti = S[SOFF_ + idx];
        S[idx] = ir; S[SOFF_ + idx] = ii;              // prefix = init state of chunk c
        float qr = Gr * ir - Gi * ii + tr;
        float qi = Gr * ii + Gi * ir + ti;
        ir = qr; ii = qi;
    }
}

__global__ __launch_bounds__(256)
void scan_final(const float* __restrict__ Bu, const float* __restrict__ Apar,
                const float* __restrict__ S, unsigned short* __restrict__ Obf) {
    int p = threadIdx.x;
    int b = blockIdx.x / CH_;
    int c = blockIdx.x % CH_;
    float Ar = Apar[p], Ai = Apar[P_ + p];
    size_t idx = ((size_t)c * B_ + b) * P_ + p;
    float sr = S[idx], si = S[SOFF_ + idx];
    int t0 = c * CL_;
    for (int t = t0; t < t0 + CL_; ++t) {
        const float* row = Bu + ((size_t)t * B_ + b) * 512;
        float br = row[p], bi = row[256 + p];
        float qr = Ar * sr - Ai * si + br;
        float qi = Ar * si + Ai * sr + bi;
        sr = qr; si = qi;
        unsigned short* orow = Obf + ((size_t)t * B_ + b) * 512;
        orow[p]       = f2bf(sr);
        orow[256 + p] = f2bf(si);
    }
}

// ---------------- host launcher ----------------
extern "C" void kernel_launch(void* const* d_in, const int* in_sizes, int n_in,
                              void* d_out, int out_size, void* d_ws, size_t ws_size,
                              hipStream_t stream) {
    const float* U      = (const float*)d_in[0];   // (L,B,H)
    const float* logLr  = (const float*)d_in[1];   // (P,)
    const float* Limag  = (const float*)d_in[2];   // (P,)
    const float* Btr    = (const float*)d_in[3];   // (P,H)
    const float* Bti    = (const float*)d_in[4];   // (P,H)
    const float* Cr     = (const float*)d_in[5];   // (H,P)
    const float* Ci     = (const float*)d_in[6];   // (H,P)
    const float* Dv     = (const float*)d_in[7];   // (H,)
    const float* logDt  = (const float*)d_in[8];   // (P,)
    float* out = (float*)d_out;

    // workspace carve-up (256B aligned)
    size_t off = 0;
    auto take = [&](size_t bytes) { size_t o = off; off = (off + bytes + 255) & ~(size_t)255; return o; };
    size_t o_apar = take(6 * P_ * sizeof(float));
    size_t o_w1   = take((size_t)512 * H_ * sizeof(unsigned short));   // W1t [512][128]
    size_t o_w2   = take((size_t)H_ * 512 * sizeof(unsigned short));   // W2t [128][512]
    size_t o_ubf  = take(M_ * H_ * sizeof(unsigned short));
    size_t o_bu   = take(M_ * 512 * sizeof(float));
    size_t o_s    = take(2 * SOFF_ * sizeof(float));
    size_t o_obf  = take(M_ * 512 * sizeof(unsigned short));
    if (ws_size < off) return;   // scratch too small -> bail deterministically

    char* ws = (char*)d_ws;
    float*          Apar = (float*)(ws + o_apar);
    unsigned short* W1t  = (unsigned short*)(ws + o_w1);
    unsigned short* W2t  = (unsigned short*)(ws + o_w2);
    unsigned short* Ubf  = (unsigned short*)(ws + o_ubf);
    float*          Bu   = (float*)(ws + o_bu);
    float*          S    = (float*)(ws + o_s);
    unsigned short* Obf  = (unsigned short*)(ws + o_obf);

    // 1) parameters
    setup_params<<<1, 256, 0, stream>>>(logLr, Limag, logDt, Apar);
    // 2) operand packing (both W matrices stored [N][K], k contiguous)
    build_w1<<<(P_ * H_) / 256, 256, 0, stream>>>(Btr, Bti, Apar, W1t);
    build_w2<<<(H_ * 512) / 256, 256, 0, stream>>>(Cr, Ci, W2t);
    convert_u<<<(int)((M_ * H_ / 4) / 256), 256, 0, stream>>>((const float4*)U, Ubf, M_ * H_ / 4);
    // 3) GEMM1: Bu[M,512] = Ubf[M,128] * W1t^T
    gemm_bf16_wmma<<<dim3((int)(M_ / BM), 512 / BN), 256, 0, stream>>>(
        Ubf, W1t, Bu, (int)M_, 512, H_, nullptr, nullptr);
    // 4) chunked scan: o[t] = A*o[t-1] + Bu[t]
    scan_local  <<<B_ * CH_, 256, 0, stream>>>(Bu, Apar, S);
    scan_combine<<<(B_ * P_) / 256, 256, 0, stream>>>(Apar, S);
    scan_final  <<<B_ * CH_, 256, 0, stream>>>(Bu, Apar, S, Obf);
    // 5) GEMM2 + epilogue: out[M,128] = Obf[M,512] * W2t^T + D*u
    gemm_bf16_wmma<<<dim3((int)(M_ / BM), H_ / BN), 256, 0, stream>>>(
        Obf, W2t, out, (int)M_, H_, 512, Dv, U);
}
